// GraphEmbv_7791070675734
// MI455X (gfx1250) — compile-verified, matrix operand order checked by reading, not run on hardware
//
#include <hip/hip_runtime.h>
#include <math.h>

// ---------------- WMMA types ----------------
typedef __attribute__((ext_vector_type(16))) __bf16 v16bf;
typedef __attribute__((ext_vector_type(8)))  __bf16 bhalf8;
typedef __attribute__((ext_vector_type(8)))  float  v8f;

#define EPS   1e-8f
#define BNEPS 1e-5f
#define LNEPS 1e-5f

__device__ __forceinline__ v16bf load_frag(const __bf16* row, int half)
{
    bhalf8 lo = *(const bhalf8*)(row + half * 8);
    bhalf8 hi = *(const bhalf8*)(row + 16 + half * 8);
    return __builtin_shufflevector(lo, hi,
        0,1,2,3,4,5,6,7,8,9,10,11,12,13,14,15);
}

// ---------------------------------------------------------------------------
// WMMA GEMM: C[M,N] = act( opA(A1,A2)[M,K] @ opB(B)[K,N] + bias )
// AMODE: 0 -> A = A1 ; 1 -> A = (A1-A2)^2 (A2 same lda) ; 2 -> A = (A1-vec)^2
// TRANSB: false -> B[k*ldb+n] ; true -> B[n*ldb+k]
// ACT: 0 none, 1 tanh
// MB: M bounds checks. KB: K bounds (chunk level; callers ensure K % 8 == 0).
// Block: 256 threads = 8 waves. Block tile 64(M) x 128(N), K step 32,
// double-buffered LDS, one barrier per K step. Wave grid 2(M) x 4(N); each
// wave owns a 32x32 output: 2 A-frags x 2 B-frags -> 4 WMMAs per K step.
// ---------------------------------------------------------------------------
template<int AMODE, bool TRANSB, int ACT, bool MB, bool KB>
__global__ __launch_bounds__(256)
void k_gemm(const float* __restrict__ A1, const float* __restrict__ A2,
            const float* __restrict__ B,  const float* __restrict__ bias,
            float* __restrict__ C,
            int M, int N, int K, int lda, int ldb, int ldc)
{
    constexpr int BM = 64, BN = 128, BK = 32;
    __shared__ __attribute__((aligned(16))) __bf16 As[2][BM * BK]; // [m][k]
    __shared__ __attribute__((aligned(16))) __bf16 Bs[2][BN * BK]; // [n][k]

    const int tid  = threadIdx.x;
    const int bm   = blockIdx.y * BM;
    const int bn   = blockIdx.x * BN;
    const int wave = tid >> 5;
    const int lane = tid & 31;
    const int wr   = (wave & 1) * 32;   // wave M offset
    const int wc   = (wave >> 1) * 32;  // wave N offset
    const int half = lane >> 4;
    const int l15  = lane & 15;

    v8f a00 = {}, a01 = {}, a10 = {}, a11 = {};
    const int nk = (K + BK - 1) / BK;

    auto fill = [&](int buf, int k0) {
        // ---- A tile: row r = tid/4, 8-float chunk col = (tid&3)*8 ----
        {
            const int r   = tid >> 2;
            const int col = (tid & 3) * 8;
            float x[8];
            bool ok = (!MB || (bm + r) < M) && (!KB || (k0 + col) < K);
            if (ok) {
                const float* p = A1 + (long)(bm + r) * lda + k0 + col;
                __builtin_prefetch(p + BK, 0, 0);       // next K panel
                float4 p0 = *(const float4*)p;
                float4 p1 = *(const float4*)(p + 4);
                x[0]=p0.x; x[1]=p0.y; x[2]=p0.z; x[3]=p0.w;
                x[4]=p1.x; x[5]=p1.y; x[6]=p1.z; x[7]=p1.w;
                if constexpr (AMODE == 1) {
                    const float* q = A2 + (long)(bm + r) * lda + k0 + col;
                    float4 q0 = *(const float4*)q;
                    float4 q1 = *(const float4*)(q + 4);
                    float y[8] = {q0.x,q0.y,q0.z,q0.w,q1.x,q1.y,q1.z,q1.w};
                    #pragma unroll
                    for (int i = 0; i < 8; ++i) { float d = x[i]-y[i]; x[i] = d*d; }
                } else if constexpr (AMODE == 2) {
                    const float* q = A2 + k0 + col;
                    float4 q0 = *(const float4*)q;
                    float4 q1 = *(const float4*)(q + 4);
                    float y[8] = {q0.x,q0.y,q0.z,q0.w,q1.x,q1.y,q1.z,q1.w};
                    #pragma unroll
                    for (int i = 0; i < 8; ++i) { float d = x[i]-y[i]; x[i] = d*d; }
                }
            } else {
                #pragma unroll
                for (int i = 0; i < 8; ++i) x[i] = 0.f;
            }
            bhalf8 h;
            #pragma unroll
            for (int i = 0; i < 8; ++i) h[i] = (__bf16)x[i];
            *(bhalf8*)&As[buf][r * BK + col] = h;
        }
        // ---- B tile into [n][k] ----
        if constexpr (TRANSB) {
            // row n = tid/2, 16-float run col = (tid&1)*16 (two 8-chunks)
            const int n    = tid >> 1;
            const int colb = (tid & 1) * 16;
            #pragma unroll
            for (int c8 = 0; c8 < 2; ++c8) {
                const int col = colb + c8 * 8;
                float x[8];
                bool ok = (!KB || (k0 + col) < K);
                if (ok) {
                    const float* p = B + (long)(bn + n) * ldb + k0 + col;
                    if (c8 == 0) __builtin_prefetch(p + BK, 0, 0);
                    float4 p0 = *(const float4*)p;
                    float4 p1 = *(const float4*)(p + 4);
                    x[0]=p0.x; x[1]=p0.y; x[2]=p0.z; x[3]=p0.w;
                    x[4]=p1.x; x[5]=p1.y; x[6]=p1.z; x[7]=p1.w;
                } else {
                    #pragma unroll
                    for (int i = 0; i < 8; ++i) x[i] = 0.f;
                }
                bhalf8 h;
                #pragma unroll
                for (int i = 0; i < 8; ++i) h[i] = (__bf16)x[i];
                *(bhalf8*)&Bs[buf][n * BK + col] = h;
            }
        } else {
            // row k = tid/8, 16-float run nbase = (tid&7)*16
            const int k     = tid >> 3;
            const int nbase = (tid & 7) * 16;
            float x[16];
            bool ok = (!KB || (k0 + k) < K);
            if (ok) {
                const float* p = B + (long)(k0 + k) * ldb + bn + nbase;
                __builtin_prefetch(p + (long)BK * ldb, 0, 0);   // next K panel
                float4 p0 = *(const float4*)p;
                float4 p1 = *(const float4*)(p + 4);
                float4 p2 = *(const float4*)(p + 8);
                float4 p3 = *(const float4*)(p + 12);
                x[0]=p0.x;  x[1]=p0.y;  x[2]=p0.z;  x[3]=p0.w;
                x[4]=p1.x;  x[5]=p1.y;  x[6]=p1.z;  x[7]=p1.w;
                x[8]=p2.x;  x[9]=p2.y;  x[10]=p2.z; x[11]=p2.w;
                x[12]=p3.x; x[13]=p3.y; x[14]=p3.z; x[15]=p3.w;
            } else {
                #pragma unroll
                for (int i = 0; i < 16; ++i) x[i] = 0.f;
            }
            #pragma unroll
            for (int i = 0; i < 16; ++i)
                Bs[buf][(nbase + i) * BK + k] = (__bf16)x[i];
        }
    };

    fill(0, 0);

    for (int kt = 0; kt < nk; ++kt) {
        __syncthreads();                       // tile kt ready; old tile free
        if (kt + 1 < nk) fill((kt + 1) & 1, (kt + 1) * BK);

        const __bf16* Ab = As[kt & 1];
        const __bf16* Bb = Bs[kt & 1];
        v16bf af0 = load_frag(&Ab[(wr + l15) * BK],      half);
        v16bf af1 = load_frag(&Ab[(wr + 16 + l15) * BK], half);
        v16bf bf0 = load_frag(&Bb[(wc + l15) * BK],      half);
        v16bf bf1 = load_frag(&Bb[(wc + 16 + l15) * BK], half);

        a00 = __builtin_amdgcn_wmma_f32_16x16x32_bf16(
                  false, af0, false, bf0, (short)0, a00, false, false);
        a01 = __builtin_amdgcn_wmma_f32_16x16x32_bf16(
                  false, af0, false, bf1, (short)0, a01, false, false);
        a10 = __builtin_amdgcn_wmma_f32_16x16x32_bf16(
                  false, af1, false, bf0, (short)0, a10, false, false);
        a11 = __builtin_amdgcn_wmma_f32_16x16x32_bf16(
                  false, af1, false, bf1, (short)0, a11, false, false);
    }

    // ---- epilogue: D layout -> M = r + 8*(lane>=16), N = lane&15 ----
    #pragma unroll
    for (int j = 0; j < 2; ++j) {
        const int n = bn + wc + j * 16 + l15;
        const float bj = bias ? bias[n] : 0.f;
        #pragma unroll
        for (int i = 0; i < 2; ++i) {
            const v8f& acc = (j == 0) ? (i == 0 ? a00 : a10)
                                      : (i == 0 ? a01 : a11);
            #pragma unroll
            for (int r = 0; r < 8; ++r) {
                int m = bm + wr + i * 16 + r + (half << 3);
                if (!MB || m < M) {
                    float v = acc[r] + bj;
                    if constexpr (ACT == 1) v = tanhf(v);
                    C[(long)m * ldc + n] = v;
                }
            }
        }
    }
}

// ---------------------------------------------------------------------------
// Small helper kernels
// ---------------------------------------------------------------------------
__global__ void k_rowmean(const float* __restrict__ in, float* __restrict__ out,
                          int R, int D)
{
    int o = blockIdx.x;
    for (int d = threadIdx.x; d < D; d += blockDim.x) {
        float s = 0.f;
        for (int r = 0; r < R; ++r) s += in[((long)o * R + r) * D + d];
        out[(long)o * D + d] = s / (float)R;
    }
}

__global__ __launch_bounds__(256)
void k_bnstats(const float* __restrict__ h, float* __restrict__ mu,
               float* __restrict__ var)
{
    __shared__ float ss[256], sq[256];
    int r = blockIdx.x, t = threadIdx.x;
    float s = 0.f, q = 0.f;
    for (int i = t; i < 64 * 1024; i += 256) {
        int b = i >> 10, d = i & 1023;
        float v = h[((long)(b * 36 + r)) * 1024 + d];
        s += v; q += v * v;
    }
    ss[t] = s; sq[t] = q; __syncthreads();
    for (int st = 128; st > 0; st >>= 1) {
        if (t < st) { ss[t] += ss[t + st]; sq[t] += sq[t + st]; }
        __syncthreads();
    }
    if (t == 0) {
        float m = ss[0] / 65536.f;
        mu[r] = m; var[r] = sq[0] / 65536.f - m * m;
    }
}

__global__ void k_bnapply_tanh(float* __restrict__ h, const float* __restrict__ mu,
                               const float* __restrict__ var,
                               const float* __restrict__ lg, const float* __restrict__ lb)
{
    int row = blockIdx.x, r = row % 36;
    float m = mu[r], iv = rsqrtf(var[r] + BNEPS), g = lg[r], bb = lb[r];
    for (int d = threadIdx.x; d < 1024; d += blockDim.x) {
        long idx = (long)row * 1024 + d;
        h[idx] = tanhf((h[idx] - m) * iv * g + bb);
    }
}

__global__ void k_bn_batch_tanh(float* __restrict__ g, const float* __restrict__ gg,
                                const float* __restrict__ gb)
{
    int d = blockIdx.x * blockDim.x + threadIdx.x;
    if (d >= 1024) return;
    float s = 0.f, q = 0.f;
    for (int b = 0; b < 64; ++b) { float v = g[(long)b * 1024 + d]; s += v; q += v * v; }
    float m = s / 64.f;
    float iv = rsqrtf(q / 64.f - m * m + BNEPS);
    for (int b = 0; b < 64; ++b) {
        float v = g[(long)b * 1024 + d];
        g[(long)b * 1024 + d] = tanhf((v - m) * iv * gg[d] + gb[d]);
    }
}

__global__ __launch_bounds__(256)
void k_rowdot3(const float* __restrict__ X, const float* __restrict__ Y,
               const float* __restrict__ v, const float* __restrict__ cb,
               float* __restrict__ out, int rpo, int D)
{
    __shared__ float red[256];
    int row = blockIdx.x, t = threadIdx.x;
    const float* x = X + (long)row * D;
    const float* y = Y + (long)(row / rpo) * D;
    float s = 0.f;
    for (int d = t; d < D; d += 256) s += x[d] * y[d] * v[d];
    red[t] = s; __syncthreads();
    for (int st = 128; st > 0; st >>= 1) {
        if (t < st) red[t] += red[t + st];
        __syncthreads();
    }
    if (t == 0) out[row] = red[0] + cb[0];
}

__global__ void k_softmax_small(const float* __restrict__ in, float* __restrict__ out,
                                int n, float scale)
{
    __shared__ float buf[64];
    int o = blockIdx.x, t = threadIdx.x;
    buf[t] = (t < n) ? in[(long)o * n + t] : -1e30f;
    __syncthreads();
    float mx = -1e30f;
    for (int i = 0; i < n; ++i) mx = fmaxf(mx, buf[i]);
    float sum = 0.f;
    for (int i = 0; i < n; ++i) sum += expf(scale * (buf[i] - mx));
    if (t < n) out[(long)o * n + t] = expf(scale * (buf[t] - mx)) / sum;
}

__global__ __launch_bounds__(256)
void k_weighted_l2(const float* __restrict__ w, const float* __restrict__ feats,
                   float* __restrict__ out, float* __restrict__ norms, int n)
{
    __shared__ float red[256];
    int o = blockIdx.x, t = threadIdx.x;
    float vals[4]; float q = 0.f;
    #pragma unroll
    for (int u = 0; u < 4; ++u) {
        int d = t + u * 256;
        float s = 0.f;
        for (int j = 0; j < n; ++j)
            s += w[(long)o * n + j] * feats[((long)o * n + j) * 1024 + d];
        vals[u] = s; q += s * s;
    }
    red[t] = q; __syncthreads();
    for (int st = 128; st > 0; st >>= 1) {
        if (t < st) red[t] += red[t + st];
        __syncthreads();
    }
    float nrm = sqrtf(red[0]) + EPS;
    #pragma unroll
    for (int u = 0; u < 4; ++u)
        out[(long)o * 1024 + t + u * 256] = vals[u] / nrm;
    if (norms && t == 0) norms[o] = nrm;
}

__global__ __launch_bounds__(256)
void k_img_par(const float* __restrict__ img, const int* __restrict__ adjs,
               const float* __restrict__ wraw, const float* __restrict__ norms,
               float* __restrict__ out)
{
    int row = blockIdx.x;
    int b = row / 36, r = row % 36;
    const int* arow = adjs + ((long)b * 36 + r) * 36;
    int eff[5]; int cnt = 0;
    for (int j = 0; j < 36 && cnt < 5; ++j)
        if (arow[j] == 1) eff[cnt++] = j;
    for (; cnt < 5; ++cnt) eff[cnt] = r;
    float wv[5]; float mx = -1e30f;
    #pragma unroll
    for (int k = 0; k < 5; ++k) { wv[k] = wraw[b * 36 + eff[k]]; mx = fmaxf(mx, wv[k]); }
    float sum = 0.f;
    #pragma unroll
    for (int k = 0; k < 5; ++k) { wv[k] = expf(wv[k] - mx); sum += wv[k]; }
    float inv = 1.f / (sum * norms[b]);
    for (int d = threadIdx.x; d < 1024; d += 256) {
        float s = 0.f;
        #pragma unroll
        for (int k = 0; k < 5; ++k)
            s += wv[k] * img[((long)b * 36 + eff[k]) * 1024 + d];
        out[(long)row * 1024 + d] = s * inv;
    }
}

__global__ void k_attn_norm_r(float* __restrict__ attn)
{
    int gid = blockIdx.x * blockDim.x + threadIdx.x;
    if (gid >= 64 * 1280) return;
    int b = gid / 1280, j = gid % 1280;
    float ss = 0.f;
    for (int r = 0; r < 36; ++r) {
        float v = attn[((long)(b * 36 + r)) * 1280 + j];
        v = v > 0.f ? v : 0.1f * v;
        ss += v * v;
    }
    float inv = 1.f / (sqrtf(ss) + EPS);
    for (int r = 0; r < 36; ++r) {
        long idx = ((long)(b * 36 + r)) * 1280 + j;
        float v = attn[idx];
        v = v > 0.f ? v : 0.1f * v;
        attn[idx] = v * inv;
    }
}

__global__ void k_attn_softmax_w(float* __restrict__ attn)
{
    int gid = blockIdx.x * blockDim.x + threadIdx.x;
    if (gid >= 2304 * 32) return;
    int row = gid / 32, c = gid % 32;
    float* p = attn + (long)row * 1280 + c * 40;
    float mx = -1e30f;
    for (int w = 0; w < 40; ++w) mx = fmaxf(mx, p[w]);
    float s = 0.f;
    for (int w = 0; w < 40; ++w) s += expf(9.f * (p[w] - mx));
    float inv = 1.f / s;
    for (int w = 0; w < 40; ++w) p[w] = expf(9.f * (p[w] - mx)) * inv;
}

__global__ __launch_bounds__(256)
void k_l2norm_rows(float* __restrict__ X, int D, int relu)
{
    __shared__ float red[256];
    int row = blockIdx.x, t = threadIdx.x;
    float q = 0.f;
    for (int d = t; d < D; d += 256) {
        float v = X[(long)row * D + d];
        if (relu) v = fmaxf(v, 0.f);
        q += v * v;
    }
    red[t] = q; __syncthreads();
    for (int st = 128; st > 0; st >>= 1) {
        if (t < st) red[t] += red[t + st];
        __syncthreads();
    }
    float inv = 1.f / (sqrtf(red[0]) + EPS);
    for (int d = t; d < D; d += 256) {
        float v = X[(long)row * D + d];
        if (relu) v = fmaxf(v, 0.f);
        X[(long)row * D + d] = v * inv;
    }
}

__global__ __launch_bounds__(256)
void k_simglo_epi(float* __restrict__ pre, float* __restrict__ simglo,
                  float* __restrict__ out, int c)
{
    __shared__ float red[256];
    int b = blockIdx.x, s = threadIdx.x;
    float v = fmaxf(pre[(long)b * 256 + s], 0.f);
    red[s] = v * v; __syncthreads();
    for (int st = 128; st > 0; st >>= 1) {
        if (s < st) red[s] += red[s + st];
        __syncthreads();
    }
    v *= 1.f / (sqrtf(red[0]) + EPS);
    simglo[(long)b * 256 + s] = v;
    long sec2 = 32L * 64 * 37 * 256;
    out[sec2 + ((long)c * 64 + b) * 256 + s] = v;
    out[(((long)c * 64 + b) * 37 + 0) * 256 + s] = v;
}

__global__ __launch_bounds__(256)
void k_wts(const float* __restrict__ simpar, const float* __restrict__ simglo,
           const float* __restrict__ simw, const float* __restrict__ simb,
           float* __restrict__ wtsraw)
{
    __shared__ float red[256];
    int row = blockIdx.x, t = threadIdx.x, b = row / 36;
    red[t] = simpar[(long)row * 256 + t] * simglo[(long)b * 256 + t] * simw[t];
    __syncthreads();
    for (int st = 128; st > 0; st >>= 1) {
        if (t < st) red[t] += red[t + st];
        __syncthreads();
    }
    if (t == 0) wtsraw[row] = red[0] + simb[0];
}

__global__ void k_layernorm_sig(const float* __restrict__ wtsraw,
                                const float* __restrict__ lng,
                                const float* __restrict__ lnb,
                                float* __restrict__ wts)
{
    __shared__ float buf[64];
    int b = blockIdx.x, t = threadIdx.x;
    buf[t] = (t < 36) ? wtsraw[b * 36 + t] : 0.f;
    __syncthreads();
    float m = 0.f;
    for (int i = 0; i < 36; ++i) m += buf[i];
    m /= 36.f;
    float vv = 0.f;
    for (int i = 0; i < 36; ++i) { float d = buf[i] - m; vv += d * d; }
    vv /= 36.f;
    if (t < 36) {
        float z = (buf[t] - m) * rsqrtf(vv + LNEPS) * lng[t] + lnb[t];
        wts[b * 36 + t] = 1.f / (1.f + expf(-z));
    }
}

__global__ __launch_bounds__(256)
void k_scale_scatter(const float* __restrict__ simloc, const float* __restrict__ wts,
                     float* __restrict__ out, int c)
{
    int row = blockIdx.x, t = threadIdx.x;
    int b = row / 36, r = row % 36;
    float wv = wts[row];
    long obase = (((long)c * 64 + b) * 37 + (1 + r)) * 256;
    out[obase + t] = wv * simloc[(long)row * 256 + t];
}

// ---------------------------------------------------------------------------
extern "C" void kernel_launch(void* const* d_in, const int* in_sizes, int n_in,
                              void* d_out, int out_size, void* d_ws, size_t ws_size,
                              hipStream_t stream)
{
    (void)in_sizes; (void)n_in; (void)out_size; (void)ws_size;
    const float* img    = (const float*)d_in[0];
    const float* cap    = (const float*)d_in[1];
    const int*   adjs   = (const int*)  d_in[3];
    const float* tsa_lw = (const float*)d_in[5];
    const float* tsa_lb = (const float*)d_in[6];
    const float* tsa_gw = (const float*)d_in[7];
    const float* tsa_gb = (const float*)d_in[8];
    const float* tsa_cw = (const float*)d_in[9];
    const float* tsa_cb = (const float*)d_in[10];
    const float* vl_w   = (const float*)d_in[11];
    const float* vl_b   = (const float*)d_in[12];
    const float* vbn_lg = (const float*)d_in[13];
    const float* vbn_lb = (const float*)d_in[14];
    const float* vg_w   = (const float*)d_in[15];
    const float* vg_b   = (const float*)d_in[16];
    const float* vbn_gg = (const float*)d_in[17];
    const float* vbn_gb = (const float*)d_in[18];
    const float* vc_w   = (const float*)d_in[19];
    const float* vc_b   = (const float*)d_in[20];
    const float* loc_w  = (const float*)d_in[21];
    const float* loc_b  = (const float*)d_in[22];
    const float* par_w  = (const float*)d_in[23];
    const float* par_b  = (const float*)d_in[24];
    const float* glo_w  = (const float*)d_in[25];
    const float* glo_b  = (const float*)d_in[26];
    const float* sim_w  = (const float*)d_in[27];
    const float* sim_b  = (const float*)d_in[28];
    const float* ln_g   = (const float*)d_in[29];
    const float* ln_b   = (const float*)d_in[30];
    float* out = (float*)d_out;
    float* ws  = (float*)d_ws;

    size_t o = 0;
    float* H      = ws + o; o += 2304L * 1024;
    float* IMGAVE = ws + o; o += 64L * 1024;
    float* G      = ws + o; o += 64L * 1024;
    float* MU     = ws + o; o += 36;
    float* VAR    = ws + o; o += 36;
    float* WRAW   = ws + o; o += 2304;
    float* WSOFT  = ws + o; o += 2304;
    float* IMGGLO = ws + o; o += 64L * 1024;
    float* NORMG  = ws + o; o += 64;
    float* IMGPAR = ws + o; o += 2304L * 1024;
    float* CAPAVE = ws + o; o += 32L * 1024;
    float* LE     = ws + o; o += 1280L * 1024;
    float* GE     = ws + o; o += 32L * 1024;
    float* CWR    = ws + o; o += 1280;
    float* CWS    = ws + o; o += 1280;
    float* CAPGLO = ws + o; o += 32L * 1024;
    float* ATTN   = ws + o; o += 2304L * 1280;
    float* CTX    = ws + o; o += 2304L * 1024;
    float* SIMLOC = ws + o; o += 2304L * 256;
    float* SIMPAR = ws + o; o += 2304L * 256;
    float* SIMGLO = ws + o; o += 64L * 256;
    float* WTSR   = ws + o; o += 2304;
    float* WTS    = ws + o; o += 2304;

    // ================= image side =================
    k_rowmean<<<64, 256, 0, stream>>>(img, IMGAVE, 36, 1024);
    k_gemm<0,false,0,false,false><<<dim3(8, 36), 256, 0, stream>>>(
        img, nullptr, vl_w, vl_b, H, 2304, 1024, 1024, 1024, 1024, 1024);
    k_bnstats<<<36, 256, 0, stream>>>(H, MU, VAR);
    k_bnapply_tanh<<<2304, 256, 0, stream>>>(H, MU, VAR, vbn_lg, vbn_lb);
    k_gemm<0,false,0,false,false><<<dim3(8, 1), 256, 0, stream>>>(
        IMGAVE, nullptr, vg_w, vg_b, G, 64, 1024, 1024, 1024, 1024, 1024);
    k_bn_batch_tanh<<<4, 256, 0, stream>>>(G, vbn_gg, vbn_gb);
    k_rowdot3<<<2304, 256, 0, stream>>>(H, G, vc_w, vc_b, WRAW, 36, 1024);
    k_softmax_small<<<64, 64, 0, stream>>>(WRAW, WSOFT, 36, 1.f);
    k_weighted_l2<<<64, 256, 0, stream>>>(WSOFT, img, IMGGLO, NORMG, 36);
    k_img_par<<<2304, 256, 0, stream>>>(img, adjs, WRAW, NORMG, IMGPAR);

    // ================= caption side =================
    k_rowmean<<<32, 256, 0, stream>>>(cap, CAPAVE, 40, 1024);
    k_gemm<0,false,1,false,false><<<dim3(8, 20), 256, 0, stream>>>(
        cap, nullptr, tsa_lw, tsa_lb, LE, 1280, 1024, 1024, 1024, 1024, 1024);
    k_gemm<0,false,1,true,false><<<dim3(8, 1), 256, 0, stream>>>(
        CAPAVE, nullptr, tsa_gw, tsa_gb, GE, 32, 1024, 1024, 1024, 1024, 1024);
    k_rowdot3<<<1280, 256, 0, stream>>>(LE, GE, tsa_cw, tsa_cb, CWR, 40, 1024);
    k_softmax_small<<<32, 64, 0, stream>>>(CWR, CWS, 40, 1.f);
    k_weighted_l2<<<32, 256, 0, stream>>>(CWS, cap, CAPGLO, nullptr, 40);

    // attn(2304,1280) = img @ cap_all^T
    k_gemm<0,true,0,false,false><<<dim3(10, 36), 256, 0, stream>>>(
        img, nullptr, cap, nullptr, ATTN, 2304, 1280, 1024, 1024, 1024, 1280);
    k_attn_norm_r<<<320, 256, 0, stream>>>(ATTN);
    k_attn_softmax_w<<<288, 256, 0, stream>>>(ATTN);

    // ================= per-caption tail =================
    for (int c = 0; c < 32; ++c) {
        const float* cap_c = cap + (long)c * 40 * 1024;
        const float* cg_c  = CAPGLO + (long)c * 1024;
        // ctx = attnS_c(2304,40) @ cap_c(40,1024) : K bounds only (K=40, %8==0)
        k_gemm<0,false,0,false,true><<<dim3(8, 36), 256, 0, stream>>>(
            ATTN + c * 40, nullptr, cap_c, nullptr, CTX,
            2304, 1024, 40, 1280, 1024, 1024);
        k_l2norm_rows<<<2304, 256, 0, stream>>>(CTX, 1024, 0);
        // sim_loc = relu((ctx-img)^2 @ loc_w + loc_b) -> l2norm
        k_gemm<1,false,0,false,false><<<dim3(2, 36), 256, 0, stream>>>(
            CTX, img, loc_w, loc_b, SIMLOC, 2304, 256, 1024, 1024, 256, 256);
        k_l2norm_rows<<<2304, 256, 0, stream>>>(SIMLOC, 256, 1);
        // sim_par = relu((img_par-cap_glo)^2 @ par_w + par_b) -> l2norm
        k_gemm<2,false,0,false,false><<<dim3(2, 36), 256, 0, stream>>>(
            IMGPAR, cg_c, par_w, par_b, SIMPAR, 2304, 256, 1024, 1024, 256, 256);
        k_l2norm_rows<<<2304, 256, 0, stream>>>(SIMPAR, 256, 1);
        // sim_glo = relu((img_glo-cap_glo)^2 @ glo_w + glo_b) -> l2norm (+out)
        k_gemm<2,false,0,false,false><<<dim3(2, 1), 256, 0, stream>>>(
            IMGGLO, cg_c, glo_w, glo_b, SIMGLO, 64, 256, 1024, 1024, 256, 256);
        k_simglo_epi<<<64, 256, 0, stream>>>(SIMGLO, SIMGLO, out, c);
        k_wts<<<2304, 256, 0, stream>>>(SIMPAR, SIMGLO, sim_w, sim_b, WTSR);
        k_layernorm_sig<<<64, 64, 0, stream>>>(WTSR, ln_g, ln_b, WTS);
        k_scale_scatter<<<2304, 256, 0, stream>>>(SIMLOC, WTS, out, c);
    }
}
